// CCMModel_17102559773377
// MI455X (gfx1250) — matrix operand order repaired and checked by reading, not compiled, not run on hardware
//
#include <hip/hip_runtime.h>
#include <hip/hip_bf16.h>
#include <math.h>

// ---------------- shapes ----------------
#define B_    64
#define PL_   128
#define RL_   64
#define TD_   63        // decode steps = RL-1
#define NT_   50
#define NP_   20
#define VW_   32000
#define DE_   300
#define TE_   100
#define GH_   512
#define G3_   1536      // 3*GH
#define KENC  512       // DE+2*TE = 500 padded to 512
#define KDEC  1632      // D_DEC = 1612 padded to 1632
#define DDEC  1612

typedef __attribute__((ext_vector_type(16))) __bf16 bfx16;
typedef __attribute__((ext_vector_type(8)))  float  fx8;
typedef __attribute__((ext_vector_type(4)))  unsigned int u32x4;
typedef __attribute__((ext_vector_type(8)))  unsigned int u32x8;

union FragU { bfx16 v; unsigned int u[8]; };

__device__ inline float sigm(float x) { return 1.0f / (1.0f + expf(-x)); }

// A fragment: 16x32 (MxK), row-major source, lda elements.
// lane l: half h=l>>4, m=mbase+(l&15). v[0..3]: K=h*16+{0..7}; v[4..7]: K=h*16+8+{0..7}
__device__ inline bfx16 loadA(const __hip_bfloat16* A, int lda, int mbase, int kbase, int lane) {
  FragU f;
  int h = lane >> 4;
  const __hip_bfloat16* row = A + (size_t)(mbase + (lane & 15)) * lda + kbase + h * 16;
  #pragma unroll
  for (int j = 0; j < 4; ++j) {
    f.u[j]     = *(const unsigned int*)(row + 2 * j);
    f.u[4 + j] = *(const unsigned int*)(row + 8 + 2 * j);
  }
  return f.v;
}

// B fragment: 32x16 (KxN) from weight stored row-major (N x K), ldb elements.
// lane l: half h=l>>4, n=nbase+(l&15). v[j]: K=h*16+{2j,2j+1}
__device__ inline bfx16 loadB(const __hip_bfloat16* W, int ldb, int nbase, int kbase, int lane) {
  FragU f;
  int h = lane >> 4;
  const __hip_bfloat16* row = W + (size_t)(nbase + (lane & 15)) * ldb + kbase + h * 16;
  #pragma unroll
  for (int j = 0; j < 8; ++j) f.u[j] = *(const unsigned int*)(row + 2 * j);
  return f.v;
}

#define WMMA_BF16(a, b, c) \
  __builtin_amdgcn_wmma_f32_16x16x32_bf16(false, (a), false, (b), (short)0, (c), false, false)

// TDM: issue a 2D tile load (tile_dim0=32 elems contiguous K, tile_dim1=256 rows,
// row stride KDEC elems, bf16) from global into LDS at lds_off. D# per ISA 8.3/8.4.
__device__ inline void tdm_load_tile(unsigned long long ga, unsigned lds_off) {
  u32x4 g0;
  g0[0] = 1u;                                             // count=1, user desc
  g0[1] = lds_off;                                        // LDS byte address
  g0[2] = (unsigned)ga;                                   // global addr [31:0]
  g0[3] = ((unsigned)(ga >> 32) & 0x01FFFFFFu) | 0x80000000u; // addr[56:32] | type=2
  u32x8 g1;
  g1[0] = 1u << 16;                                       // data_size = 2 bytes
  g1[1] = ((unsigned)KDEC & 0xFFFFu) << 16;               // tensor_dim0 lo16 (atomic addr 0)
  g1[2] = (((unsigned)VW_ & 0xFFFFu) << 16) | ((unsigned)KDEC >> 16); // dim1 lo16 | dim0 hi16
  g1[3] = (32u << 16) | ((unsigned)VW_ >> 16);            // tile_dim0=32 | dim1 hi16
  g1[4] = 256u;                                           // tile_dim1=256, tile_dim2=0
  g1[5] = (unsigned)KDEC;                                 // tensor_dim0_stride lo32
  g1[6] = 0u;                                             // stride0 hi16 | stride1 lo16
  g1[7] = 0u;
  asm volatile("tensor_load_to_lds %0, %1" :: "s"(g0), "s"(g1) : "memory");
}

// ---------------- utility kernels ----------------

__global__ void cvt_pad(const float* __restrict__ src, __hip_bfloat16* __restrict__ dst,
                        int rows, int sc, int dc) {
  size_t total = (size_t)rows * dc;
  for (size_t i = (size_t)blockIdx.x * blockDim.x + threadIdx.x; i < total;
       i += (size_t)gridDim.x * blockDim.x) {
    int r = (int)(i / dc), c = (int)(i % dc);
    float v = (c < sc) ? src[(size_t)r * sc + c] : 0.0f;
    dst[i] = __float2bfloat16(v);
  }
}

__global__ void zero_f4(float4* p, size_t n4) {
  for (size_t i = (size_t)blockIdx.x * blockDim.x + threadIdx.x; i < n4;
       i += (size_t)gridDim.x * blockDim.x)
    p[i] = make_float4(0.f, 0.f, 0.f, 0.f);
}

__global__ void gather_tef(const float* __restrict__ transe, const int* __restrict__ triple,
                           float* __restrict__ tef) {
  size_t total = (size_t)B_ * NT_ * NP_ * 3 * TE_;
  for (size_t i = (size_t)blockIdx.x * blockDim.x + threadIdx.x; i < total;
       i += (size_t)gridDim.x * blockDim.x) {
    int d = (int)(i % TE_);
    size_t e = i / TE_;
    int idx = triple[e];
    tef[i] = transe[(size_t)idx * TE_ + d];
  }
}

__global__ void triple_score(const float* __restrict__ tef, const int* __restrict__ triple,
                             const float* wh_w, const float* wh_b,
                             const float* wr_w, const float* wr_b,
                             const float* wt_w, const float* wt_b,
                             float* __restrict__ sl) {
  int bnp = blockIdx.x;
  __shared__ float t3[3 * TE_];
  __shared__ float red[128];
  const float* base = tef + (size_t)bnp * (3 * TE_);
  for (int i = threadIdx.x; i < 3 * TE_; i += 128) t3[i] = base[i];
  __syncthreads();
  float val = 0.f;
  int i = threadIdx.x;
  if (i < TE_) {
    float hp = wh_b[i], rp = wr_b[i], tp = wt_b[i];
    for (int j = 0; j < TE_; ++j) {
      hp += wh_w[i * TE_ + j] * t3[j];
      rp += wr_w[i * TE_ + j] * t3[TE_ + j];
      tp += wt_w[i * TE_ + j] * t3[2 * TE_ + j];
    }
    val = rp * tanhf(hp + tp);
  }
  red[threadIdx.x] = val;
  __syncthreads();
  for (int s = 64; s > 0; s >>= 1) {
    if (threadIdx.x < s) red[threadIdx.x] += red[threadIdx.x + s];
    __syncthreads();
  }
  if (threadIdx.x == 0) {
    float s = red[0];
    if (triple[(size_t)bnp * 3] == 0) s = -1000000000.0f;
    sl[bnp] = s;
  }
}

__global__ void static_graph_k(const float* __restrict__ tef, const float* __restrict__ sl,
                               float* __restrict__ sg) {
  int bn = blockIdx.x;
  __shared__ float sa[NP_];
  if (threadIdx.x == 0) {
    float m = -1e30f;
    for (int p = 0; p < NP_; ++p) m = fmaxf(m, sl[(size_t)bn * NP_ + p]);
    float s = 0.f;
    for (int p = 0; p < NP_; ++p) { float e = expf(sl[(size_t)bn * NP_ + p] - m); sa[p] = e; s += e; }
    float inv = 1.f / s;
    for (int p = 0; p < NP_; ++p) sa[p] *= inv;
  }
  __syncthreads();
  for (int d = threadIdx.x; d < 2 * TE_; d += 256) {
    int off = (d < TE_) ? d : (TE_ + d);
    float acc = 0.f;
    for (int p = 0; p < NP_; ++p)
      acc += sa[p] * tef[((size_t)bn * NP_ + p) * (3 * TE_) + off];
    sg[(size_t)bn * (2 * TE_) + d] = acc;
  }
}

__global__ void ubpre_k(const float* __restrict__ sg, const float* ub_w, const float* ub_b,
                        float* __restrict__ ubpre) {
  int bn = blockIdx.x;
  int i = threadIdx.x;
  if (i < TE_) {
    float a = ub_b[i];
    const float* g = sg + (size_t)bn * (2 * TE_);
    for (int d = 0; d < 2 * TE_; ++d) a += ub_w[i * (2 * TE_) + d] * g[d];
    ubpre[(size_t)bn * TE_ + i] = a;
  }
}

__global__ void build_postin(const float* __restrict__ word_emb, const float* __restrict__ sg,
                             const int* __restrict__ post, const int* __restrict__ post_triple,
                             __hip_bfloat16* __restrict__ pin) {
  int bp = blockIdx.x;
  int b = bp >> 7;
  int w = post[bp];
  int tr = post_triple[bp];
  __hip_bfloat16* dst = pin + (size_t)bp * KENC;
  for (int c = threadIdx.x; c < KENC; c += 256) {
    float v;
    if (c < DE_)            v = word_emb[(size_t)w * DE_ + c];
    else if (c < DE_ + 200) v = sg[((size_t)b * NT_ + tr) * 200 + (c - DE_)];
    else                    v = 0.f;
    dst[c] = __float2bfloat16(v);
  }
}

__global__ void build_respfeat(const float* __restrict__ word_emb, const float* __restrict__ transe,
                               const int* __restrict__ response, const int* __restrict__ rtrip,
                               __hip_bfloat16* __restrict__ ofeat) {
  int m = blockIdx.x;
  int b = m / TD_, t = m % TD_;
  int w = response[b * RL_ + t];
  const int* rt = rtrip + ((size_t)b * RL_ + t) * 3;
  __hip_bfloat16* dst = ofeat + (size_t)m * KDEC;
  for (int c = threadIdx.x; c < 620; c += 256) {
    float v;
    if (c < 300)      v = word_emb[(size_t)w * DE_ + c];
    else if (c < 600) { int e = (c - 300) / TE_, d = (c - 300) % TE_;
                        v = transe[(size_t)rt[e] * TE_ + d]; }
    else              v = 0.f;
    dst[1012 + c] = __float2bfloat16(v);
  }
}

// ---------------- generic bf16 WMMA GEMM (unroll-by-2 software pipeline) ----------------
// block = 256 threads = 8 waves (2x4), block tile 32x256, wave tile 16x64.
// Two named fragment sets alternate across the unrolled halves: no register rotation.
__global__ void gemm_bf16(const __hip_bfloat16* __restrict__ A, const __hip_bfloat16* __restrict__ W,
                          float* __restrict__ C, const float* __restrict__ bias,
                          int M, int N, int K, int lda, int ldb, int ldc) {
  int wave = threadIdx.x >> 5, lane = threadIdx.x & 31;
  int mbase = blockIdx.y * 32 + (wave >> 2) * 16;
  int n0 = blockIdx.x * 256 + (wave & 3) * 64;
  if (mbase >= M || n0 >= N) return;
  fx8 acc[4] = {};
  bfx16 a0 = loadA(A, lda, mbase, 0, lane), a1;
  bfx16 b0[4], b1[4];
  #pragma unroll
  for (int t = 0; t < 4; ++t) b0[t] = loadB(W, ldb, n0 + 16 * t, 0, lane);
  int k = 0;
  for (; k + 64 <= K; k += 64) {
    // fill set 1 (chunk k+32) while set 0 (chunk k) computes
    a1 = loadA(A, lda, mbase, k + 32, lane);
    #pragma unroll
    for (int t = 0; t < 4; ++t) b1[t] = loadB(W, ldb, n0 + 16 * t, k + 32, lane);
    #pragma unroll
    for (int t = 0; t < 4; ++t) acc[t] = WMMA_BF16(a0, b0[t], acc[t]);
    // fill set 0 (chunk k+64) while set 1 computes
    if (k + 64 < K) {
      a0 = loadA(A, lda, mbase, k + 64, lane);
      #pragma unroll
      for (int t = 0; t < 4; ++t) b0[t] = loadB(W, ldb, n0 + 16 * t, k + 64, lane);
    }
    #pragma unroll
    for (int t = 0; t < 4; ++t) acc[t] = WMMA_BF16(a1, b1[t], acc[t]);
  }
  if (k < K) {  // K % 64 == 32 tail: set 0 holds chunk k
    #pragma unroll
    for (int t = 0; t < 4; ++t) acc[t] = WMMA_BF16(a0, b0[t], acc[t]);
  }
  int hh = lane >> 4, nn = lane & 15;
  #pragma unroll
  for (int t = 0; t < 4; ++t) {
    int n = n0 + 16 * t + nn;
    float bv = bias ? bias[n] : 0.f;
    #pragma unroll
    for (int r = 0; r < 8; ++r) {
      int m = mbase + hh * 8 + r;
      C[(size_t)m * ldc + n] = acc[t][r] + bv;
    }
  }
}

// ---------------- final vocab GEMM: TDM double-buffered LDS for B tiles ----------------
// M=4032 (rows m=b*63+t), N=32000, K=1632. Block tile 32x256, 8 waves.
// Wave 0 DMAs the shared 256x32 bf16 B tile into LDS via tensor_load_to_lds,
// ping-pong buffered, synchronized with TENSORcnt + block barrier.
__global__ void gemm_out_tdm(const __hip_bfloat16* __restrict__ A,
                             const __hip_bfloat16* __restrict__ W,
                             const float* __restrict__ bias, float* __restrict__ out) {
  __shared__ unsigned int Bsm[2][256 * 16];   // 2 x 16KB (256 rows x 32 bf16)
  int wave = threadIdx.x >> 5, lane = threadIdx.x & 31;
  int mbase = blockIdx.y * 32 + (wave >> 2) * 16;
  int n0blk = blockIdx.x * 256;
  int wn = wave & 3;
  const int NK = KDEC / 32;                   // 51
  const __hip_bfloat16* Wblk = W + (size_t)n0blk * KDEC;

  if (wave == 0)
    tdm_load_tile((unsigned long long)(size_t)Wblk, (unsigned)(size_t)&Bsm[0][0]);

  fx8 acc[4] = {};
  bfx16 a = loadA(A, KDEC, mbase, 0, lane);
  for (int kk = 0; kk < NK; ++kk) {
    if (wave == 0) {
      if (kk + 1 < NK) {
        tdm_load_tile((unsigned long long)(size_t)(Wblk + (kk + 1) * 32),
                      (unsigned)(size_t)&Bsm[(kk + 1) & 1][0]);
        __builtin_amdgcn_s_wait_tensorcnt(1);   // in-order: tile kk now resident
      } else {
        __builtin_amdgcn_s_wait_tensorcnt(0);
      }
    }
    __syncthreads();
    bfx16 ac = a;
    if (kk + 1 < NK) a = loadA(A, KDEC, mbase, (kk + 1) * 32, lane);  // overlap with ds+wmma
    int h = lane >> 4;
    #pragma unroll
    for (int t = 0; t < 4; ++t) {
      FragU f;
      int row = wn * 64 + 16 * t + (lane & 15);
      #pragma unroll
      for (int j = 0; j < 8; ++j) f.u[j] = Bsm[kk & 1][row * 16 + h * 8 + j];
      acc[t] = WMMA_BF16(ac, f.v, acc[t]);
    }
    __syncthreads();
  }
  int hh = lane >> 4, nn = lane & 15;
  #pragma unroll
  for (int t = 0; t < 4; ++t) {
    int v = n0blk + wn * 64 + 16 * t + nn;
    float bv = bias[v];
    #pragma unroll
    for (int r = 0; r < 8; ++r) {
      int m = mbase + hh * 8 + r;
      int b = m / TD_, tt = m % TD_;
      out[((size_t)b * VW_ + v) * TD_ + tt] = acc[t][r] + bv;
    }
  }
}

// ---------------- encoder step: gh = h@Whh^T fused with GRU gates ----------------
__global__ void enc_step(const __hip_bfloat16* __restrict__ hbf_in, const float* __restrict__ h_in,
                         float* __restrict__ h_out, __hip_bfloat16* __restrict__ hbf_out,
                         const float* __restrict__ gx, const __hip_bfloat16* __restrict__ Whh,
                         const float* __restrict__ bhh, const int* __restrict__ post_length,
                         float* __restrict__ pout, int t) {
  int wave = threadIdx.x >> 5, lane = threadIdx.x & 31;
  int mbase = wave * 16;
  int j0 = blockIdx.x * 16;
  fx8 gh[3];
  #pragma unroll
  for (int g = 0; g < 3; ++g) {
    fx8 c = {};
    for (int k = 0; k < GH_; k += 32) {
      bfx16 a = loadA(hbf_in, GH_, mbase, k, lane);
      bfx16 b = loadB(Whh, GH_, g * GH_ + j0, k, lane);
      c = WMMA_BF16(a, b, c);
    }
    gh[g] = c;
  }
  int hh = lane >> 4, jn = j0 + (lane & 15);
  #pragma unroll
  for (int r = 0; r < 8; ++r) {
    int b = mbase + hh * 8 + r;
    const float* gxr = gx + ((size_t)b * PL_ + t) * G3_;
    float ghr = gh[0][r] + bhh[jn];
    float ghz = gh[1][r] + bhh[GH_ + jn];
    float ghn = gh[2][r] + bhh[2 * GH_ + jn];
    float rr = sigm(gxr[jn] + ghr);
    float zz = sigm(gxr[GH_ + jn] + ghz);
    float nn = tanhf(gxr[2 * GH_ + jn] + rr * ghn);
    float hp = h_in[b * GH_ + jn];
    float hcand = (1.f - zz) * nn + zz * hp;
    bool valid = t < post_length[b];
    float hnew = valid ? hcand : hp;
    h_out[b * GH_ + jn] = hnew;
    hbf_out[b * GH_ + jn] = __float2bfloat16(hnew);
    pout[((size_t)b * PL_ + t) * GH_ + jn] = valid ? hcand : 0.f;
  }
}

// ---------------- decoder per-step kernels ----------------

__global__ void dec_attn(const float* __restrict__ h, const float* __restrict__ pout,
                         const __hip_bfloat16* __restrict__ ofeat,
                         __hip_bfloat16* __restrict__ decin, int t) {
  int b = blockIdx.x;
  __shared__ float hs[GH_];
  __shared__ float sc[PL_];
  for (int j = threadIdx.x; j < GH_; j += 256) hs[j] = h[b * GH_ + j];
  __syncthreads();
  if (threadIdx.x < PL_) {
    const float* row = pout + ((size_t)b * PL_ + threadIdx.x) * GH_;
    float s = 0.f;
    for (int j = 0; j < GH_; ++j) s += row[j] * hs[j];
    sc[threadIdx.x] = s;
  }
  __syncthreads();
  if (threadIdx.x == 0) {
    float m = -1e30f;
    for (int p = 0; p < PL_; ++p) m = fmaxf(m, sc[p]);
    float s = 0.f;
    for (int p = 0; p < PL_; ++p) { sc[p] = expf(sc[p] - m); s += sc[p]; }
    float inv = 1.f / s;
    for (int p = 0; p < PL_; ++p) sc[p] *= inv;
  }
  __syncthreads();
  for (int j = threadIdx.x; j < GH_; j += 256) {
    float acc = 0.f;
    for (int p = 0; p < PL_; ++p) acc += sc[p] * pout[((size_t)b * PL_ + p) * GH_ + j];
    decin[(size_t)b * KDEC + j] = __float2bfloat16(acc);
  }
  const __hip_bfloat16* src = ofeat + (size_t)(b * TD_ + t) * KDEC;
  for (int c = threadIdx.x; c < 600; c += 256)
    decin[(size_t)b * KDEC + 1012 + c] = src[1012 + c];
}

__global__ void dec_graph(const float* __restrict__ h, const float* __restrict__ ubpre,
                          const float* __restrict__ sg,
                          const float* wb_w, const float* wb_b,
                          const float* vb_w, const float* vb_b,
                          float* __restrict__ da, __hip_bfloat16* __restrict__ decin,
                          __hip_bfloat16* __restrict__ ofeat, int t) {
  int b = blockIdx.x;
  __shared__ float hs[GH_];
  __shared__ float wbh[TE_];
  __shared__ float dl[NT_];
  for (int j = threadIdx.x; j < GH_; j += 256) hs[j] = h[b * GH_ + j];
  __syncthreads();
  if (threadIdx.x < TE_) {
    float a = wb_b[threadIdx.x];
    for (int j = 0; j < GH_; ++j) a += wb_w[threadIdx.x * GH_ + j] * hs[j];
    wbh[threadIdx.x] = a;
  }
  __syncthreads();
  if (threadIdx.x < NT_) {
    float a = vb_b[0];
    const float* up = ubpre + ((size_t)b * NT_ + threadIdx.x) * TE_;
    for (int i = 0; i < TE_; ++i) a += vb_w[i] * tanhf(wbh[i] + up[i]);
    dl[threadIdx.x] = a;
  }
  __syncthreads();
  if (threadIdx.x == 0) {
    float m = -1e30f;
    for (int n = 0; n < NT_; ++n) m = fmaxf(m, dl[n]);
    float s = 0.f;
    for (int n = 0; n < NT_; ++n) { dl[n] = expf(dl[n] - m); s += dl[n]; }
    float inv = 1.f / s;
    for (int n = 0; n < NT_; ++n) dl[n] *= inv;
  }
  __syncthreads();
  if (threadIdx.x < NT_) da[b * NT_ + threadIdx.x] = dl[threadIdx.x];
  for (int d = threadIdx.x; d < 2 * TE_; d += 256) {
    float a = 0.f;
    for (int n = 0; n < NT_; ++n) a += dl[n] * sg[((size_t)b * NT_ + n) * (2 * TE_) + d];
    __hip_bfloat16 v = __float2bfloat16(a);
    decin[(size_t)b * KDEC + GH_ + d] = v;
    ofeat[(size_t)(b * TD_ + t) * KDEC + GH_ + d] = v;
  }
}

__global__ void dec_triple(const float* __restrict__ h, const float* __restrict__ tef,
                           const float* __restrict__ da,
                           const float* wc_w, const float* wc_b,
                           __hip_bfloat16* __restrict__ decin,
                           __hip_bfloat16* __restrict__ ofeat, int t) {
  int b = blockIdx.x;
  __shared__ float hs[GH_];
  __shared__ float q[3 * TE_];
  __shared__ float acc[3 * TE_];
  __shared__ float ta[NP_];
  for (int j = threadIdx.x; j < GH_; j += 256) hs[j] = h[b * GH_ + j];
  __syncthreads();
  for (int i = threadIdx.x; i < 3 * TE_; i += 256) {
    float a = wc_b[i];
    for (int j = 0; j < GH_; ++j) a += wc_w[i * GH_ + j] * hs[j];
    q[i] = a;
    acc[i] = 0.f;
  }
  __syncthreads();
  for (int n = 0; n < NT_; ++n) {
    if (threadIdx.x < NP_) {
      const float* row = tef + (((size_t)b * NT_ + n) * NP_ + threadIdx.x) * (3 * TE_);
      float s = 0.f;
      for (int d = 0; d < 3 * TE_; ++d) s += row[d] * q[d];
      ta[threadIdx.x] = s;
    }
    __syncthreads();
    if (threadIdx.x == 0) {
      float m = -1e30f;
      for (int p = 0; p < NP_; ++p) m = fmaxf(m, ta[p]);
      float s = 0.f;
      for (int p = 0; p < NP_; ++p) { ta[p] = expf(ta[p] - m); s += ta[p]; }
      float inv = 1.f / s;
      for (int p = 0; p < NP_; ++p) ta[p] *= inv;
    }
    __syncthreads();
    float w = da[b * NT_ + n];
    for (int d = threadIdx.x; d < 3 * TE_; d += 256) {
      float p = 0.f;
      for (int pp = 0; pp < NP_; ++pp)
        p += ta[pp] * tef[(((size_t)b * NT_ + n) * NP_ + pp) * (3 * TE_) + d];
      acc[d] += w * p;
    }
    __syncthreads();
  }
  for (int d = threadIdx.x; d < 3 * TE_; d += 256) {
    __hip_bfloat16 v = __float2bfloat16(acc[d]);
    decin[(size_t)b * KDEC + 712 + d] = v;
    ofeat[(size_t)(b * TD_ + t) * KDEC + 712 + d] = v;
  }
}

__global__ void dec_gru(const __hip_bfloat16* __restrict__ decin,
                        const __hip_bfloat16* __restrict__ hbf_in, const float* __restrict__ h_in,
                        const __hip_bfloat16* __restrict__ Wihd, const __hip_bfloat16* __restrict__ Whhd,
                        const float* __restrict__ bih, const float* __restrict__ bhh,
                        float* __restrict__ h_out, __hip_bfloat16* __restrict__ hbf_out,
                        __hip_bfloat16* __restrict__ ofeat, int t) {
  int wave = threadIdx.x >> 5, lane = threadIdx.x & 31;
  int mbase = wave * 16;
  int j0 = blockIdx.x * 16;
  fx8 gi[3], gh[3];
  #pragma unroll
  for (int g = 0; g < 3; ++g) {
    fx8 c = {};
    for (int k = 0; k < KDEC; k += 32) {
      bfx16 a = loadA(decin, KDEC, mbase, k, lane);
      bfx16 b = loadB(Wihd, KDEC, g * GH_ + j0, k, lane);
      c = WMMA_BF16(a, b, c);
    }
    gi[g] = c;
    fx8 c2 = {};
    for (int k = 0; k < GH_; k += 32) {
      bfx16 a = loadA(hbf_in, GH_, mbase, k, lane);
      bfx16 b = loadB(Whhd, GH_, g * GH_ + j0, k, lane);
      c2 = WMMA_BF16(a, b, c2);
    }
    gh[g] = c2;
  }
  int hh = lane >> 4, jn = j0 + (lane & 15);
  #pragma unroll
  for (int r = 0; r < 8; ++r) {
    int b = mbase + hh * 8 + r;
    float rr = sigm(gi[0][r] + bih[jn] + gh[0][r] + bhh[jn]);
    float zz = sigm(gi[1][r] + bih[GH_ + jn] + gh[1][r] + bhh[GH_ + jn]);
    float nn = tanhf(gi[2][r] + bih[2 * GH_ + jn] + rr * (gh[2][r] + bhh[2 * GH_ + jn]));
    float hp = h_in[b * GH_ + jn];
    float hn = (1.f - zz) * nn + zz * hp;
    h_out[b * GH_ + jn] = hn;
    hbf_out[b * GH_ + jn] = __float2bfloat16(hn);
    ofeat[(size_t)(b * TD_ + t) * KDEC + jn] = __float2bfloat16(hn);
  }
}

// ---------------- host ----------------
static inline size_t alignup(size_t x) { return (x + 255) & ~(size_t)255; }
static inline int gblocks(size_t n, int tpb) {
  size_t b = (n + tpb - 1) / tpb;
  return (int)(b < 32768 ? b : 32768);
}

extern "C" void kernel_launch(void* const* d_in, const int* in_sizes, int n_in,
                              void* d_out, int out_size, void* d_ws, size_t ws_size,
                              hipStream_t stream) {
  const float* word_emb  = (const float*)d_in[0];
  const float* transe    = (const float*)d_in[1];
  const float* wh_w = (const float*)d_in[2];  const float* wh_b = (const float*)d_in[3];
  const float* wr_w = (const float*)d_in[4];  const float* wr_b = (const float*)d_in[5];
  const float* wt_w = (const float*)d_in[6];  const float* wt_b = (const float*)d_in[7];
  const float* wb_w = (const float*)d_in[8];  const float* wb_b = (const float*)d_in[9];
  const float* ub_w = (const float*)d_in[10]; const float* ub_b = (const float*)d_in[11];
  const float* vb_w = (const float*)d_in[12]; const float* vb_b = (const float*)d_in[13];
  const float* wc_w = (const float*)d_in[14]; const float* wc_b = (const float*)d_in[15];
  const float* Wih_e = (const float*)d_in[16]; const float* Whh_e = (const float*)d_in[17];
  const float* bih_e = (const float*)d_in[18]; const float* bhh_e = (const float*)d_in[19];
  const float* Wih_d = (const float*)d_in[20]; const float* Whh_d = (const float*)d_in[21];
  const float* bih_d = (const float*)d_in[22]; const float* bhh_d = (const float*)d_in[23];
  const float* out_w = (const float*)d_in[24]; const float* out_b = (const float*)d_in[25];
  const int* post        = (const int*)d_in[26];
  const int* post_length = (const int*)d_in[27];
  const int* response    = (const int*)d_in[28];
  const int* rtrip       = (const int*)d_in[29];
  const int* post_triple = (const int*)d_in[30];
  const int* triple      = (const int*)d_in[31];
  float* out = (float*)d_out;

  char* p = (char*)d_ws;
  #define WSALLOC(ty, name, elems) ty* name = (ty*)p; p += alignup((size_t)(elems) * sizeof(ty))
  WSALLOC(float,          tef,     (size_t)B_ * NT_ * NP_ * 3 * TE_);
  WSALLOC(float,          sl,      (size_t)B_ * NT_ * NP_);
  WSALLOC(float,          sg,      (size_t)B_ * NT_ * 2 * TE_);
  WSALLOC(float,          ubpre,   (size_t)B_ * NT_ * TE_);
  WSALLOC(__hip_bfloat16, pin,     (size_t)B_ * PL_ * KENC);
  WSALLOC(__hip_bfloat16, wihe_bf, (size_t)G3_ * KENC);
  WSALLOC(__hip_bfloat16, whhe_bf, (size_t)G3_ * GH_);
  WSALLOC(__hip_bfloat16, wihd_bf, (size_t)G3_ * KDEC);
  WSALLOC(__hip_bfloat16, whhd_bf, (size_t)G3_ * GH_);
  WSALLOC(__hip_bfloat16, outw_bf, (size_t)VW_ * KDEC);
  WSALLOC(float,          gxe,     (size_t)B_ * PL_ * G3_);
  WSALLOC(float,          pout,    (size_t)B_ * PL_ * GH_);
  WSALLOC(float,          hf,      (size_t)2 * B_ * GH_);
  WSALLOC(__hip_bfloat16, hbf,     (size_t)2 * B_ * GH_);
  WSALLOC(__hip_bfloat16, decin,   (size_t)B_ * KDEC);
  WSALLOC(float,          da,      (size_t)B_ * NT_);
  WSALLOC(__hip_bfloat16, ofeat,   (size_t)B_ * TD_ * KDEC);
  #undef WSALLOC
  (void)ws_size; (void)n_in; (void)in_sizes; (void)out_size;

  cvt_pad<<<gblocks((size_t)VW_ * KDEC, 256), 256, 0, stream>>>(out_w, outw_bf, VW_, DDEC, KDEC);
  cvt_pad<<<gblocks((size_t)G3_ * KDEC, 256), 256, 0, stream>>>(Wih_d, wihd_bf, G3_, DDEC, KDEC);
  cvt_pad<<<gblocks((size_t)G3_ * KENC, 256), 256, 0, stream>>>(Wih_e, wihe_bf, G3_, 500, KENC);
  cvt_pad<<<gblocks((size_t)G3_ * GH_, 256), 256, 0, stream>>>(Whh_e, whhe_bf, G3_, GH_, GH_);
  cvt_pad<<<gblocks((size_t)G3_ * GH_, 256), 256, 0, stream>>>(Whh_d, whhd_bf, G3_, GH_, GH_);

  {
    size_t bytes = ((char*)da) - ((char*)hf);
    size_t n4 = bytes / 16;
    zero_f4<<<gblocks(n4, 256), 256, 0, stream>>>((float4*)hf, n4);
  }

  gather_tef<<<gblocks((size_t)B_ * NT_ * NP_ * 3 * TE_, 256), 256, 0, stream>>>(transe, triple, tef);
  triple_score<<<B_ * NT_ * NP_, 128, 0, stream>>>(tef, triple, wh_w, wh_b, wr_w, wr_b, wt_w, wt_b, sl);
  static_graph_k<<<B_ * NT_, 256, 0, stream>>>(tef, sl, sg);
  ubpre_k<<<B_ * NT_, 128, 0, stream>>>(sg, ub_w, ub_b, ubpre);

  build_postin<<<B_ * PL_, 256, 0, stream>>>(word_emb, sg, post, post_triple, pin);
  gemm_bf16<<<dim3(G3_ / 256, (B_ * PL_) / 32), 256, 0, stream>>>(
      pin, wihe_bf, gxe, bih_e, B_ * PL_, G3_, KENC, KENC, KENC, G3_);
  build_respfeat<<<B_ * TD_, 256, 0, stream>>>(word_emb, transe, response, rtrip, ofeat);

  for (int t = 0; t < PL_; ++t) {
    int cur = t & 1, nxt = cur ^ 1;
    enc_step<<<GH_ / 16, 128, 0, stream>>>(
        hbf + (size_t)cur * B_ * GH_, hf + (size_t)cur * B_ * GH_,
        hf + (size_t)nxt * B_ * GH_, hbf + (size_t)nxt * B_ * GH_,
        gxe, whhe_bf, bhh_e, post_length, pout, t);
  }

  for (int s = 0; s < TD_; ++s) {
    int cur = s & 1, nxt = cur ^ 1;
    const float* hc = hf + (size_t)cur * B_ * GH_;
    dec_attn<<<B_, 256, 0, stream>>>(hc, pout, ofeat, decin, s);
    dec_graph<<<B_, 256, 0, stream>>>(hc, ubpre, sg, wb_w, wb_b, vb_w, vb_b, da, decin, ofeat, s);
    dec_triple<<<B_, 256, 0, stream>>>(hc, tef, da, wc_w, wc_b, decin, ofeat, s);
    dec_gru<<<GH_ / 16, 128, 0, stream>>>(
        decin, hbf + (size_t)cur * B_ * GH_, hc,
        wihd_bf, whhd_bf, bih_d, bhh_d,
        hf + (size_t)nxt * B_ * GH_, hbf + (size_t)nxt * B_ * GH_, ofeat, s);
  }

  gemm_out_tdm<<<dim3(VW_ / 256, (B_ * TD_) / 32), 256, 0, stream>>>(ofeat, outw_bf, out_b, out);
}
